// DecoderTranslator_56925496541198
// MI455X (gfx1250) — matrix operand order repaired and checked by reading, not compiled, not run on hardware
//
#include <hip/hip_runtime.h>
#include <hip/hip_bf16.h>
#include <math.h>

typedef __attribute__((ext_vector_type(16))) __bf16 v16bf;
typedef __attribute__((ext_vector_type(8)))  __bf16 v8bf;
typedef __attribute__((ext_vector_type(8)))  float  v8f;

#define BB   32
#define TT   128
#define SS   128
#define HID  512
#define EMB  256
#define VOC  32000
#define G4   2048   // 4*HID

// ---------------------------------------------------------------------------
// Register-blocked bf16 GEMM:  C[M,N] = A[M,K] @ B[N,K]^T (+ addend) (+ bias[n])
// Each wave computes a 32x64 macro-tile = MT(2) x NT(4) WMMA tiles of 16x16,
// so each A fragment is reused 4x and each B fragment 2x:
// 12 b128 loads -> 8 v_wmma_f32_16x16x32_bf16 per K-step (vs 4 loads / 1 wmma
// in the unblocked version).  Requires: M % 32 == 0, N % 64 == 0, K % 32 == 0.
// ---------------------------------------------------------------------------
__global__ void __launch_bounds__(256)
gemm_bf16_wmma(const __bf16* __restrict__ A, int lda,
               const __bf16* __restrict__ Bm, int ldb,
               float* __restrict__ C, int ldc,
               const float* __restrict__ addend,
               const float* __restrict__ bias,
               int M, int N, int K)
{
    const int lane = threadIdx.x & 31;
    const int wid  = threadIdx.x >> 5;
    const int tilesN4 = N >> 6;                // groups of 4 col-tiles
    const int tilesM2 = M >> 5;                // groups of 2 row-tiles
    const int group = blockIdx.x * 8 + wid;
    if (group >= tilesM2 * tilesN4) return;    // wave-uniform exit, EXEC all-1s
    const int tm0 = (group / tilesN4) * 2;     // first row tile
    const int tn0 = (group % tilesN4) * 4;     // first col tile

    const int half = lane >> 4;                // 0 or 1
    const int l15  = lane & 15;

    // A fragment: lane<16 -> row M=lane, K in {0..7, 16..23}; lane>=16 -> {8..15, 24..31}
    const __bf16* pa0 = A + (size_t)(tm0 * 16 + l15) * lda + half * 8;
    const __bf16* pa1 = pa0 + (size_t)16 * lda;
    // B fragment: lane<16 -> col N=lane, K in {0..7, 16..23}; lane>=16 -> {8..15, 24..31}
    const __bf16* pb0 = Bm + (size_t)(tn0 * 16 + l15) * ldb + half * 8;

    v8f acc[2][4];
#pragma unroll
    for (int i = 0; i < 2; ++i)
#pragma unroll
        for (int j = 0; j < 4; ++j) acc[i][j] = (v8f){};

    for (int kk = 0; kk < K; kk += 32) {
        v8bf a0lo = *(const v8bf*)(pa0 + kk);
        v8bf a0hi = *(const v8bf*)(pa0 + kk + 16);
        v8bf a1lo = *(const v8bf*)(pa1 + kk);
        v8bf a1hi = *(const v8bf*)(pa1 + kk + 16);
        v16bf a0, a1;
#pragma unroll
        for (int i = 0; i < 8; ++i) {
            a0[i] = a0lo[i]; a0[8 + i] = a0hi[i];
            a1[i] = a1lo[i]; a1[8 + i] = a1hi[i];
        }
#pragma unroll
        for (int j = 0; j < 4; ++j) {
            const __bf16* pb = pb0 + (size_t)(16 * j) * ldb + kk;
            __builtin_prefetch((const void*)(pb + 128), 0, 3);   // global_prefetch_b8 (near)
            v8bf blo = *(const v8bf*)(pb);
            v8bf bhi = *(const v8bf*)(pb + 16);
            v16bf bfm;
#pragma unroll
            for (int i = 0; i < 8; ++i) { bfm[i] = blo[i]; bfm[8 + i] = bhi[i]; }
            acc[0][j] = __builtin_amdgcn_wmma_f32_16x16x32_bf16(
                            false, a0, false, bfm, (short)0, acc[0][j], false, false);
            acc[1][j] = __builtin_amdgcn_wmma_f32_16x16x32_bf16(
                            false, a1, false, bfm, (short)0, acc[1][j], false, false);
        }
    }

#pragma unroll
    for (int mt = 0; mt < 2; ++mt) {
#pragma unroll
        for (int j = 0; j < 4; ++j) {
#pragma unroll
            for (int r = 0; r < 8; ++r) {
                const int m = (tm0 + mt) * 16 + half * 8 + r;  // D layout: vgpr r -> M=half*8+r
                const int n = (tn0 + j) * 16 + l15;
                const size_t off2 = (size_t)m * ldc + n;
                float v = acc[mt][j][r];
                if (addend) v += addend[off2];
                if (bias)   v += bias[n];
                C[off2] = v;
            }
        }
    }
}

// ---------------------------------------------------------------------------
// Elementwise / small kernels
// ---------------------------------------------------------------------------
__global__ void f32_to_bf16_k(const float* __restrict__ s, __bf16* __restrict__ d, int n) {
    int i = blockIdx.x * 256 + threadIdx.x;
    if (i < n) d[i] = (__bf16)s[i];
}

__global__ void attn_split_k(const float* __restrict__ attnW,
                             __bf16* __restrict__ wh, __bf16* __restrict__ we) {
    int idx = blockIdx.x * 256 + threadIdx.x;          // 512*512
    if (idx >= HID * HID) return;
    int n = idx >> 9, k = idx & 511;
    wh[idx] = (__bf16)attnW[(size_t)n * 1024 + k];
    we[idx] = (__bf16)attnW[(size_t)n * 1024 + 512 + k];
}

__global__ void embed_gather_k(const int* __restrict__ X, const float* __restrict__ emb,
                               __bf16* __restrict__ Ebf) {
    int idx = blockIdx.x * 256 + threadIdx.x;          // (T-1)*B*EMB
    if (idx >= (TT - 1) * BB * EMB) return;
    int e = idx & (EMB - 1);
    int r = idx >> 8;                                  // t*B + b
    int b = r & 31, t = r >> 5;
    int tok = X[b * TT + t];
    Ebf[idx] = (__bf16)emb[(size_t)tok * EMB + e];
}

__global__ void bias_sum_k(const float* a, const float* b, float* o, int n) {
    int i = blockIdx.x * 256 + threadIdx.x;
    if (i < n) o[i] = a[i] + b[i];
}

__global__ void init_state_k(const float* __restrict__ hE, const float* __restrict__ cE,
                             float* h0, float* c0, float* h1, float* c1,
                             __bf16* h0b, __bf16* h1b) {
    int idx = blockIdx.x * 256 + threadIdx.x;          // B*HID
    if (idx >= BB * HID) return;
    const int BH = BB * HID;
    float a0 = 0.5f * (hE[idx] + hE[BH + idx]);
    float a1 = 0.5f * (hE[2 * BH + idx] + hE[3 * BH + idx]);
    h0[idx] = a0; h1[idx] = a1;
    h0b[idx] = (__bf16)a0; h1b[idx] = (__bf16)a1;
    c0[idx] = 0.5f * (cE[idx] + cE[BH + idx]);
    c1[idx] = 0.5f * (cE[2 * BH + idx] + cE[3 * BH + idx]);
}

__global__ void zero_out_k(float* o) { o[0] = 0.f; }

__global__ void lstm_cell_k(const float* __restrict__ gates,
                            float* __restrict__ c, float* __restrict__ h,
                            __bf16* __restrict__ hbf) {
    int idx = blockIdx.x * 256 + threadIdx.x;          // B*HID
    if (idx >= BB * HID) return;
    int b = idx >> 9, j = idx & 511;
    const float* g = gates + (size_t)b * G4;
    float gi = g[j], gf = g[512 + j], gg = g[1024 + j], go = g[1536 + j];
    float si = 1.f / (1.f + expf(-gi));
    float sf = 1.f / (1.f + expf(-gf));
    float so = 1.f / (1.f + expf(-go));
    float cn = sf * c[idx] + si * tanhf(gg);
    float hn = so * tanhf(cn);
    c[idx] = cn; h[idx] = hn; hbf[idx] = (__bf16)hn;
}

__global__ void __launch_bounds__(256)
attn_scores_k(const float* __restrict__ hqp, const float* __restrict__ encp,
              const float* __restrict__ vvec, float* __restrict__ scores) {
    int lane = threadIdx.x & 31, wid = threadIdx.x >> 5;
    int p = blockIdx.x * 8 + wid;                      // B*S pairs
    if (p >= BB * SS) return;
    int b = p >> 7;
    const float* hq = hqp + (size_t)b * HID;
    const float* ep = encp + (size_t)p * HID;
    float sum = 0.f;
    for (int h = lane; h < HID; h += 32)
        sum += tanhf(hq[h] + ep[h]) * vvec[h];
    for (int m = 16; m; m >>= 1) sum += __shfl_xor(sum, m, 32);
    if (lane == 0) scores[p] = sum;
}

__global__ void attn_softmax_k(const float* __restrict__ scores, float* __restrict__ att) {
    __shared__ float sm[SS];
    int b = blockIdx.x, s = threadIdx.x;
    float x = scores[b * SS + s];
    sm[s] = x; __syncthreads();
    for (int o = 64; o; o >>= 1) { if (s < o) sm[s] = fmaxf(sm[s], sm[s + o]); __syncthreads(); }
    float mx = sm[0]; __syncthreads();
    float e = expf(x - mx);
    sm[s] = e; __syncthreads();
    for (int o = 64; o; o >>= 1) { if (s < o) sm[s] += sm[s + o]; __syncthreads(); }
    att[b * SS + s] = e / sm[0];
}

__global__ void attn_weighted_k(const float* __restrict__ att, const float* __restrict__ enc,
                                float* __restrict__ wtd) {
    int b = blockIdx.x, h = threadIdx.x;               // 512 threads
    const float* a = att + (size_t)b * SS;
    const float* e = enc + (size_t)b * SS * HID + h;
    float s = 0.f;
    for (int t = 0; t < SS; ++t) s += a[t] * e[(size_t)t * HID];
    wtd[(size_t)b * HID + h] = s;
}

__global__ void cat_pack_k(const float* __restrict__ h1, const float* __restrict__ wtd,
                           __bf16* __restrict__ cat) {
    int idx = blockIdx.x * 256 + threadIdx.x;          // B*1024
    if (idx >= BB * 2 * HID) return;
    int b = idx >> 10, j = idx & 1023;
    float v = (j < HID) ? h1[b * HID + j] : wtd[b * HID + (j - HID)];
    cat[idx] = (__bf16)v;
}

__global__ void __launch_bounds__(256)
row_nll_k(const float* __restrict__ logits, const int* __restrict__ X, int t,
          float* __restrict__ nll) {
    __shared__ float sm[256];
    int b = blockIdx.x, tid = threadIdx.x;
    const float* row = logits + (size_t)b * VOC;
    float mx = -3.4e38f;
    for (int i = tid; i < VOC; i += 256) mx = fmaxf(mx, row[i]);
    sm[tid] = mx; __syncthreads();
    for (int o = 128; o; o >>= 1) { if (tid < o) sm[tid] = fmaxf(sm[tid], sm[tid + o]); __syncthreads(); }
    mx = sm[0]; __syncthreads();
    float s = 0.f;
    for (int i = tid; i < VOC; i += 256) s += expf(row[i] - mx);
    sm[tid] = s; __syncthreads();
    for (int o = 128; o; o >>= 1) { if (tid < o) sm[tid] += sm[tid + o]; __syncthreads(); }
    if (tid == 0) {
        int y = X[b * TT + t + 1];
        nll[b] = mx + logf(sm[0]) - row[y];
    }
}

__global__ void step_loss_k(const float* __restrict__ nll, const int* __restrict__ X,
                            int t, float* __restrict__ out) {
    int lane = threadIdx.x;                            // 32 threads
    int y = X[lane * TT + t + 1];
    float valid = (y != 0) ? 1.f : 0.f;
    float s = nll[lane] * valid;
    float c = valid;
    for (int m = 16; m; m >>= 1) { s += __shfl_xor(s, m, 32); c += __shfl_xor(c, m, 32); }
    if (lane == 0) out[0] += s / fmaxf(c, 1.f) / (float)(TT - 1);
}

// ---------------------------------------------------------------------------
// Host driver
// ---------------------------------------------------------------------------
extern "C" void kernel_launch(void* const* d_in, const int* in_sizes, int n_in,
                              void* d_out, int out_size, void* d_ws, size_t ws_size,
                              hipStream_t stream)
{
    const int*   X     = (const int*)  d_in[0];
    const float* enc   = (const float*)d_in[1];
    const float* hE    = (const float*)d_in[2];
    const float* cE    = (const float*)d_in[3];
    const float* emb   = (const float*)d_in[4];
    const float* Wih0  = (const float*)d_in[5];
    const float* Whh0  = (const float*)d_in[6];
    const float* bih0  = (const float*)d_in[7];
    const float* bhh0  = (const float*)d_in[8];
    const float* Wih1  = (const float*)d_in[9];
    const float* Whh1  = (const float*)d_in[10];
    const float* bih1  = (const float*)d_in[11];
    const float* bhh1  = (const float*)d_in[12];
    const float* attnW = (const float*)d_in[13];
    const float* attnb = (const float*)d_in[14];
    const float* vv    = (const float*)d_in[15];
    const float* fcW   = (const float*)d_in[16];
    const float* fcb   = (const float*)d_in[17];
    (void)in_sizes; (void)n_in; (void)out_size; (void)ws_size;

    char* ws = (char*)d_ws;
    size_t off = 0;
    auto alloc = [&](size_t bytes) -> char* {
        char* p = ws + off;
        off = (off + bytes + 255) & ~(size_t)255;
        return p;
    };

    __bf16* fcWb  = (__bf16*)alloc((size_t)VOC * 1024 * 2);            // 64 MB (L2-resident)
    float*  Xih0  = (float*) alloc((size_t)(TT-1) * BB * G4 * 4);      // 33 MB
    __bf16* Ebf   = (__bf16*)alloc((size_t)(TT-1) * BB * EMB * 2);
    __bf16* encb  = (__bf16*)alloc((size_t)BB * SS * HID * 2);
    float*  encp  = (float*) alloc((size_t)BB * SS * HID * 4);         // 8 MB
    __bf16* Wih0b = (__bf16*)alloc((size_t)G4 * EMB * 2);
    __bf16* Whh0b = (__bf16*)alloc((size_t)G4 * HID * 2);
    __bf16* Wih1b = (__bf16*)alloc((size_t)G4 * HID * 2);
    __bf16* Whh1b = (__bf16*)alloc((size_t)G4 * HID * 2);
    __bf16* aWhb  = (__bf16*)alloc((size_t)HID * HID * 2);
    __bf16* aWeb  = (__bf16*)alloc((size_t)HID * HID * 2);
    float*  bsum0 = (float*) alloc(G4 * 4);
    float*  bsum1 = (float*) alloc(G4 * 4);
    float*  h0f   = (float*) alloc(BB * HID * 4);
    float*  c0f   = (float*) alloc(BB * HID * 4);
    float*  h1f   = (float*) alloc(BB * HID * 4);
    float*  c1f   = (float*) alloc(BB * HID * 4);
    __bf16* h0b   = (__bf16*)alloc(BB * HID * 2);
    __bf16* h1b   = (__bf16*)alloc(BB * HID * 2);
    float*  gates = (float*) alloc(BB * G4 * 4);
    float*  hqp   = (float*) alloc(BB * HID * 4);
    float*  scor  = (float*) alloc(BB * SS * 4);
    float*  att   = (float*) alloc(BB * SS * 4);
    float*  wtd   = (float*) alloc(BB * HID * 4);
    __bf16* catb  = (__bf16*)alloc(BB * 2 * HID * 2);
    float*  logit = (float*) alloc((size_t)BB * VOC * 4);              // 4 MB
    float*  nll   = (float*) alloc(BB * 4);

    auto gemm = [&](const __bf16* A, int lda, const __bf16* B, int ldb,
                    float* C, int ldc, const float* addend, const float* bias,
                    int M, int N, int K) {
        int groups = (M / 32) * (N / 64);      // 32x64 macro-tiles per wave
        int blocks = (groups + 7) / 8;
        gemm_bf16_wmma<<<blocks, 256, 0, stream>>>(A, lda, B, ldb, C, ldc,
                                                   addend, bias, M, N, K);
    };
    auto conv = [&](const float* s, __bf16* d, int n) {
        f32_to_bf16_k<<<(n + 255) / 256, 256, 0, stream>>>(s, d, n);
    };

    // ---- one-time setup (per call; deterministic) ----
    zero_out_k<<<1, 1, 0, stream>>>((float*)d_out);
    conv(fcW,  fcWb,  VOC * 1024);
    conv(Wih0, Wih0b, G4 * EMB);
    conv(Whh0, Whh0b, G4 * HID);
    conv(Wih1, Wih1b, G4 * HID);
    conv(Whh1, Whh1b, G4 * HID);
    conv(enc,  encb,  BB * SS * HID);
    attn_split_k<<<(HID * HID + 255) / 256, 256, 0, stream>>>(attnW, aWhb, aWeb);
    embed_gather_k<<<((TT-1) * BB * EMB + 255) / 256, 256, 0, stream>>>(X, emb, Ebf);
    bias_sum_k<<<(G4 + 255) / 256, 256, 0, stream>>>(bih0, bhh0, bsum0, G4);
    bias_sum_k<<<(G4 + 255) / 256, 256, 0, stream>>>(bih1, bhh1, bsum1, G4);
    init_state_k<<<(BB * HID + 255) / 256, 256, 0, stream>>>(hE, cE, h0f, c0f, h1f, c1f, h0b, h1b);

    // Loop-invariant hoists:
    //   Xih0[t,b,:] = E @ W_ih0^T + (b_ih0 + b_hh0)      [4.3 GF once, not per step]
    gemm(Ebf, EMB, Wih0b, EMB, Xih0, G4, nullptr, bsum0, (TT-1) * BB, G4, EMB);
    //   enc_proj = enc @ attnW[:,512:]^T + attnb         [kills 4.3 GF/step of attention]
    gemm(encb, HID, aWeb, HID, encp, HID, nullptr, attnb, BB * SS, HID, HID);

    // ---- sequential scan over 127 timesteps ----
    for (int t = 0; t < TT - 1; ++t) {
        // layer 0: gates = Xih0[t] + h0 @ W_hh0^T
        gemm(h0b, HID, Whh0b, HID, gates, G4, Xih0 + (size_t)t * BB * G4, nullptr, BB, G4, HID);
        lstm_cell_k<<<(BB * HID + 255) / 256, 256, 0, stream>>>(gates, c0f, h0f, h0b);
        // layer 1: gates = h0_new @ W_ih1^T + bias; gates += h1 @ W_hh1^T
        gemm(h0b, HID, Wih1b, HID, gates, G4, nullptr, bsum1, BB, G4, HID);
        gemm(h1b, HID, Whh1b, HID, gates, G4, gates, nullptr, BB, G4, HID);
        lstm_cell_k<<<(BB * HID + 255) / 256, 256, 0, stream>>>(gates, c1f, h1f, h1b);
        // attention
        gemm(h1b, HID, aWhb, HID, hqp, HID, nullptr, nullptr, BB, HID, HID);
        attn_scores_k<<<(BB * SS + 7) / 8, 256, 0, stream>>>(hqp, encp, vv, scor);
        attn_softmax_k<<<BB, SS, 0, stream>>>(scor, att);
        attn_weighted_k<<<BB, HID, 0, stream>>>(att, enc, wtd);
        // logits = [h1, weighted] @ fcW^T + fcb   (dominant GEMM, fcW bf16 L2-resident)
        cat_pack_k<<<(BB * 2 * HID + 255) / 256, 256, 0, stream>>>(h1f, wtd, catb);
        gemm(catb, 2 * HID, fcWb, 2 * HID, logit, VOC, nullptr, fcb, BB, VOC, 2 * HID);
        // loss
        row_nll_k<<<BB, 256, 0, stream>>>(logit, X, t, nll);
        step_loss_k<<<1, 32, 0, stream>>>(nll, X, t, (float*)d_out);
    }
}
